// TopKSAE_11647951306780
// MI455X (gfx1250) — compile-verified, matrix operand order checked by reading, not run on hardware
//
#include <hip/hip_runtime.h>
#include <stdint.h>

// ---------------------------------------------------------------------------
// Types for CDNA5 WMMA / TDM
// ---------------------------------------------------------------------------
typedef __attribute__((ext_vector_type(16))) __bf16 v16bf;
typedef __attribute__((ext_vector_type(8)))  float  v8f;
typedef __attribute__((ext_vector_type(4)))  unsigned int u32x4;
typedef __attribute__((ext_vector_type(8)))  int          i32x8;
typedef __attribute__((ext_vector_type(4)))  int          i32x4;

#define TILE_M 128
#define TILE_N 128
#define TILE_K 32
#define LSTR   40   // LDS row stride (bf16 elems): 32 + 8 pad -> conflict-free b128 reads

union FragBF {
    v16bf v;
    uint4 q[2];
};

static __device__ __forceinline__ uint32_t f32u(float f) { union { float f; uint32_t u; } c; c.f = f; return c.u; }
static __device__ __forceinline__ float u32f(uint32_t u) { union { uint32_t u; float f; } c; c.u = u; return c.f; }

// round-to-nearest-even f32 -> bf16 (bit trick; inputs finite, moderate range)
static __device__ __forceinline__ uint16_t bf16_rne(float x) {
    uint32_t u = f32u(x);
    return (uint16_t)((u + 0x7FFFu + ((u >> 16) & 1u)) >> 16);
}
static __device__ __forceinline__ float bf16f(uint16_t h) { return u32f(((uint32_t)h) << 16); }

// Convert 4 f32 to (hi, lo) bf16 pairs and store 8B each into LDS
static __device__ __forceinline__ void cvt_store4(uint16_t* __restrict__ hiA,
                                                  uint16_t* __restrict__ loA,
                                                  int o, float4 v) {
    uint16_t h0 = bf16_rne(v.x), h1 = bf16_rne(v.y), h2 = bf16_rne(v.z), h3 = bf16_rne(v.w);
    uint16_t l0 = bf16_rne(v.x - bf16f(h0));
    uint16_t l1 = bf16_rne(v.y - bf16f(h1));
    uint16_t l2 = bf16_rne(v.z - bf16f(h2));
    uint16_t l3 = bf16_rne(v.w - bf16f(h3));
    uint2 hp, lp;
    hp.x = (uint32_t)h0 | ((uint32_t)h1 << 16);
    hp.y = (uint32_t)h2 | ((uint32_t)h3 << 16);
    lp.x = (uint32_t)l0 | ((uint32_t)l1 << 16);
    lp.y = (uint32_t)l2 | ((uint32_t)l3 << 16);
    *(uint2*)(hiA + o) = hp;
    *(uint2*)(loA + o) = lp;
}

// ---------------------------------------------------------------------------
// TDM: issue TENSOR_LOAD_TO_LDS of a 2D tile (tile_rows x tile_cols f32,
// row stride = row_len_elems) from global to LDS (contiguous, row-major).
// Descriptor bitfields per CDNA5 ISA 8.3-8.6 (2D: groups 2/3 zero).
// This toolchain's builtin is the 6-arg clang-23 form:
//   (uint32x4 g0, int32x8 g1, int32x4 g2, int32x4 g3, int32x8 extra, i32 cpol)
// ---------------------------------------------------------------------------
static __device__ __forceinline__ void tdm_load_tile_f32(
    uint32_t lds_byte_off, const float* gsrc,
    int row_len_elems, int nrows_total, int tile_rows, int tile_cols) {
    uint64_t ga = (uint64_t)(uintptr_t)gsrc;
    u32x4 g0;
    g0[0] = 1u;                                   // count=1, user-mode load, no gather
    g0[1] = lds_byte_off;                         // lds_addr (bytes)
    g0[2] = (uint32_t)ga;                         // global_addr[31:0]
    g0[3] = (uint32_t)((ga >> 32) & 0x1FFFFFFu)   // global_addr[56:32]
          | (2u << 30);                           // type = 2 ("image")
    i32x8 g1;
    g1[0] = (int)(2u << 16);                      // data_size=2 (4B); wg_mask=0 (no cluster)
    g1[1] = (int)(((uint32_t)row_len_elems & 0xFFFFu) << 16);          // tensor_dim0[15:0]
    g1[2] = (int)(((uint32_t)row_len_elems >> 16)
          | (((uint32_t)nrows_total & 0xFFFFu) << 16));                // dim0[31:16] | dim1[15:0]
    g1[3] = (int)(((uint32_t)nrows_total >> 16)
          | ((uint32_t)tile_cols << 16));                              // dim1[31:16] | tile_dim0
    g1[4] = (int)((uint32_t)tile_rows);           // tile_dim1 (tile_dim2 = 0 -> 2D)
    g1[5] = (int)((uint32_t)row_len_elems);       // tensor_dim0_stride[31:0]
    g1[6] = 0;                                    // dim0_stride[47:32]=0 | dim1_stride[15:0]=0
    g1[7] = 0;
    i32x4 g2 = {0, 0, 0, 0};
    i32x4 g3 = {0, 0, 0, 0};
    i32x8 g4 = {0, 0, 0, 0, 0, 0, 0, 0};          // extra group (zero-filled)
    __builtin_amdgcn_tensor_load_to_lds(g0, g1, g2, g3, g4, 0);
}

// ---------------------------------------------------------------------------
// C[M,N] = A[M,K] * B[N,K]^T (+ bias[N]), split-bf16 (hi/lo) via WMMA, f32 acc.
// TDM double-buffers raw f32 tiles into LDS; all waves convert to hi/lo bf16
// tiles (padded stride), then 8 waves x 8 N-subtiles x 3 WMMA per K-step.
// ---------------------------------------------------------------------------
__global__ __launch_bounds__(256)
void gemm_abT_splitbf16(const float* __restrict__ A, const float* __restrict__ B,
                        float* __restrict__ C, const float* __restrict__ bias,
                        int M, int N, int K, int has_bias) {
    __shared__ float    Af32[2][TILE_M * TILE_K];   // TDM staging (double buffer)
    __shared__ float    Bf32[2][TILE_N * TILE_K];
    __shared__ uint16_t Ahi[TILE_M * LSTR];
    __shared__ uint16_t Alo[TILE_M * LSTR];
    __shared__ uint16_t Bhi[TILE_N * LSTR];
    __shared__ uint16_t Blo[TILE_N * LSTR];

    const int tid  = threadIdx.x;
    const int lane = tid & 31;
    const int wave = tid >> 5;     // 0..7 -> 16-row slice of the 128-row tile
    const int r    = lane & 15;
    const int hb   = lane >> 4;    // 0/1: which K half-chunk this lane owns

    const int tile_m = blockIdx.y * TILE_M;
    const int tile_n = blockIdx.x * TILE_N;
    const float* Atile = A + (size_t)tile_m * K;
    const float* Btile = B + (size_t)tile_n * K;

    v8f acc[8];
#pragma unroll
    for (int i = 0; i < 8; ++i) acc[i] = v8f{0.f, 0.f, 0.f, 0.f, 0.f, 0.f, 0.f, 0.f};

    // Prologue: wave 0 kicks off the TDM for the first K-tile.
    if (tid < 32) {
        tdm_load_tile_f32((uint32_t)(uintptr_t)&Af32[0][0], Atile, K, M, TILE_M, TILE_K);
        tdm_load_tile_f32((uint32_t)(uintptr_t)&Bf32[0][0], Btile, K, N, TILE_N, TILE_K);
    }

    int buf = 0;
    for (int kb = 0; kb < K; kb += TILE_K) {
        if (tid < 32) {
            if (kb + TILE_K < K) {
                // Issue next tile's DMA, then wait for the current tile only:
                // 2 newly-issued ops may remain outstanding (overlap with compute).
                tdm_load_tile_f32((uint32_t)(uintptr_t)&Af32[buf ^ 1][0],
                                  Atile + kb + TILE_K, K, M, TILE_M, TILE_K);
                tdm_load_tile_f32((uint32_t)(uintptr_t)&Bf32[buf ^ 1][0],
                                  Btile + kb + TILE_K, K, N, TILE_N, TILE_K);
                __builtin_amdgcn_s_wait_tensorcnt(2);
            } else {
                __builtin_amdgcn_s_wait_tensorcnt(0);
            }
        }
        __syncthreads();   // current f32 tiles visible to all waves

        // All 256 threads convert the f32 tiles to hi/lo bf16 tiles.
        // 1024 float4 per matrix / 256 threads = 4 each; conflict-free b128.
#pragma unroll
        for (int i = 0; i < 4; ++i) {
            int f   = tid + i * 256;     // float4 id 0..1023
            int row = f >> 3;            // 8 float4 per 32-wide row
            int c4  = (f & 7) << 2;      // column (floats)
            int o   = row * LSTR + c4;
            float4 av = *(const float4*)&Af32[buf][row * TILE_K + c4];
            cvt_store4(Ahi, Alo, o, av);
            float4 bv = *(const float4*)&Bf32[buf][row * TILE_K + c4];
            cvt_store4(Bhi, Blo, o, bv);
        }
        __syncthreads();

        // Per ISA 7.12.2 (16-bit A frag): lane r holds K = hb*8+e (e<8) and
        // K = 16+hb*8+(e-8) (e>=8) -> two 16B chunks per fragment.
        const int arow = wave * 16 + r;
        FragBF a_hi, a_lo;
        a_hi.q[0] = *(const uint4*)(Ahi + arow * LSTR + hb * 8);
        a_hi.q[1] = *(const uint4*)(Ahi + arow * LSTR + hb * 8 + 16);
        a_lo.q[0] = *(const uint4*)(Alo + arow * LSTR + hb * 8);
        a_lo.q[1] = *(const uint4*)(Alo + arow * LSTR + hb * 8 + 16);

#pragma unroll
        for (int ns = 0; ns < 8; ++ns) {
            const int brow = ns * 16 + r;
            FragBF b_hi, b_lo;
            b_hi.q[0] = *(const uint4*)(Bhi + brow * LSTR + hb * 8);
            b_hi.q[1] = *(const uint4*)(Bhi + brow * LSTR + hb * 8 + 16);
            b_lo.q[0] = *(const uint4*)(Blo + brow * LSTR + hb * 8);
            b_lo.q[1] = *(const uint4*)(Blo + brow * LSTR + hb * 8 + 16);

            // a*b ~= a_hi*b_hi + a_hi*b_lo + a_lo*b_hi  (f32 accumulate)
            acc[ns] = __builtin_amdgcn_wmma_f32_16x16x32_bf16(
                false, a_hi.v, false, b_hi.v, (short)0, acc[ns], false, false);
            acc[ns] = __builtin_amdgcn_wmma_f32_16x16x32_bf16(
                false, a_hi.v, false, b_lo.v, (short)0, acc[ns], false, false);
            acc[ns] = __builtin_amdgcn_wmma_f32_16x16x32_bf16(
                false, a_lo.v, false, b_hi.v, (short)0, acc[ns], false, false);
        }
        buf ^= 1;
    }

    // Epilogue: C/D layout (ISA 7.12.2): lane 0-15 col=lane M=j ; lane 16-31 M=j+8
#pragma unroll
    for (int ns = 0; ns < 8; ++ns) {
        const int col = tile_n + ns * 16 + r;
        const float badd = has_bias ? bias[col] : 0.0f;
#pragma unroll
        for (int j = 0; j < 8; ++j) {
            const int row = tile_m + wave * 16 + hb * 8 + j;
            C[(size_t)row * N + col] = acc[ns][j] + badd;
        }
    }
}

// ---------------------------------------------------------------------------
// Top-K=32 by |value| per row of h_pre (row length HID), scatter into h_sparse.
// One 256-thread block per row. Abs values cached in LDS; 32 rounds of
// tree-argmax + single-owner rescan after removing the winner.
// ---------------------------------------------------------------------------
#define TK_T   256
#define TK_K   32

__global__ __launch_bounds__(256)
void topk_scatter(const float* __restrict__ h_pre, float* __restrict__ h_sparse,
                  int HID) {
    extern __shared__ float smem[];
    float* absv  = smem;                 // HID floats
    float* candv = absv + HID;           // TK_T
    float* redv  = candv + TK_T;         // TK_T
    int*   candi = (int*)(redv + TK_T);  // TK_T
    int*   redi  = candi + TK_T;         // TK_T

    const int row = blockIdx.x;
    const int t   = threadIdx.x;
    const int ept = HID / TK_T;
    const float* hp = h_pre + (size_t)row * HID;
    float* hs = h_sparse + (size_t)row * HID;

    float bv = -1.0f;
    int   bi = t;
    for (int j = 0; j < ept; ++j) {
        int idx = t + j * TK_T;          // coalesced
        float v = hp[idx];
        float a = fabsf(v);
        absv[idx] = a;
        hs[idx]   = 0.0f;                // zero-init output row
        if (a > bv) { bv = a; bi = idx; }
    }
    candv[t] = bv;
    candi[t] = bi;
    __syncthreads();

    for (int k = 0; k < TK_K; ++k) {
        redv[t] = candv[t];
        redi[t] = candi[t];
        __syncthreads();
#pragma unroll
        for (int s = TK_T / 2; s > 0; s >>= 1) {
            if (t < s) {
                if (redv[t + s] > redv[t]) { redv[t] = redv[t + s]; redi[t] = redi[t + s]; }
            }
            __syncthreads();
        }
        const int win = redi[0];
        if (t == (win & (TK_T - 1))) {   // owner of the winning index
            absv[win] = -1.0f;
            hs[win]   = hp[win];         // keep original signed value
            float nv = -1.0f; int ni = t;
            for (int j = 0; j < ept; ++j) {
                int idx = t + j * TK_T;
                float a = absv[idx];
                if (a > nv) { nv = a; ni = idx; }
            }
            candv[t] = nv;
            candi[t] = ni;
        }
        __syncthreads();
    }
}

// ---------------------------------------------------------------------------
// Launcher: out = [recon (M*IN) | h_sparse (M*H) | h_pre (M*H)]
// ---------------------------------------------------------------------------
extern "C" void kernel_launch(void* const* d_in, const int* in_sizes, int n_in,
                              void* d_out, int out_size, void* d_ws, size_t ws_size,
                              hipStream_t stream) {
    const float* x     = (const float*)d_in[0];  // (M, IN)
    const float* W_enc = (const float*)d_in[1];  // (H, IN)
    const float* W_dec = (const float*)d_in[2];  // (IN, H)
    const float* b_dec = (const float*)d_in[3];  // (IN,)

    const int IN = in_sizes[3];            // 1024
    const int H  = in_sizes[1] / IN;       // 16384
    const int M  = in_sizes[0] / IN;       // 4096

    float* out      = (float*)d_out;
    float* recon    = out;                                // M*IN
    float* h_sparse = out + (size_t)M * IN;               // M*H
    float* h_pre    = h_sparse + (size_t)M * H;           // M*H

    // 1) encoder: h_pre[M,H] = x[M,IN] * W_enc[H,IN]^T
    {
        dim3 grid(H / TILE_N, M / TILE_M), block(256);
        gemm_abT_splitbf16<<<grid, block, 0, stream>>>(
            x, W_enc, h_pre, nullptr, M, H, IN, 0);
    }

    // 2) top-k mask/scatter: h_sparse from h_pre
    {
        size_t shmem = (size_t)H * sizeof(float) + 2 * TK_T * sizeof(float)
                       + 2 * TK_T * sizeof(int);
        topk_scatter<<<M, TK_T, shmem, stream>>>(h_pre, h_sparse, H);
    }

    // 3) decoder: recon[M,IN] = h_sparse[M,H] * W_dec[IN,H]^T + b_dec
    {
        dim3 grid(IN / TILE_N, M / TILE_M), block(256);
        gemm_abT_splitbf16<<<grid, block, 0, stream>>>(
            h_sparse, W_dec, recon, b_dec, M, IN, H, 1);
    }
}
